// PVConv_85358180040854
// MI455X (gfx1250) — matrix operand order repaired and checked by reading, not compiled, not run on hardware
//
#include <hip/hip_runtime.h>
#include <hip/hip_bf16.h>

#define RG   32
#define BSZ  4
#define CIN_ 32
#define COUT_ 64
#define NPTS 65536
#define PADC 0.1f
#define NEGS 0.1f
#define R3   (RG*RG*RG)

typedef __attribute__((ext_vector_type(16))) __bf16       v16bf;
typedef __attribute__((ext_vector_type(8)))  float        v8f;
typedef __attribute__((ext_vector_type(8)))  unsigned int u32x8;
typedef __attribute__((ext_vector_type(4)))  unsigned int u32x4;
typedef __attribute__((ext_vector_type(8)))  int          i32x8;
typedef __attribute__((ext_vector_type(4)))  int          i32x4;

static __device__ __forceinline__ unsigned short f2bf(float f) {
  __bf16 h = (__bf16)f;
  return __builtin_bit_cast(unsigned short, h);
}
static __device__ __forceinline__ float bf2f(unsigned short u) {
  __bf16 h = __builtin_bit_cast(__bf16, u);
  return (float)h;
}
static __device__ __forceinline__ float norm_coord(float c) {
  float nc = (c / (1.0f + PADC) + 1.0f) * 0.5f * (RG - 1);
  return fminf(fmaxf(nc, 0.0f), (float)(RG - 1));
}
// 16-bit A-matrix 16x32 order packed here (contiguous k, not ISA table order): lane L -> row L%16, khalf g=L/16;
// element e of the lane's v16bf -> K = e + 16g
static __device__ __forceinline__ int kmap(int e, int g) {
  return e + 16 * g;
}

// ---- TDM: 1-D bulk copy global -> LDS (D# per ISA ch.8; data_size=8B) -------
static __device__ __forceinline__ void tdm_load_to_lds(const void* gsrc,
                                                       unsigned lds_off_bytes,
                                                       unsigned nbytes) {
  unsigned ne8 = nbytes >> 3;                         // elements of 8 bytes
  unsigned long long ga = (unsigned long long)(size_t)gsrc;
  u32x4 g0;
  g0[0] = 1u;                                         // count=1 (valid), user mode
  g0[1] = lds_off_bytes;                              // lds_addr
  g0[2] = (unsigned)(ga & 0xFFFFFFFFu);               // global_addr[31:0]
  g0[3] = (unsigned)((ga >> 32) & 0x01FFFFFFu)        // global_addr[56:32]
          | 0x80000000u;                              // type=2 ("image")
  i32x8 g1;
  g1[0] = (int)(3u << 16);                            // data_size=3 (8B), no masks
  g1[1] = (int)((ne8 & 0xFFFFu) << 16);               // tensor_dim0[15:0]
  g1[2] = (int)(((ne8 >> 16) & 0xFFFFu) | (1u << 16));// tensor_dim0[31:16] | tensor_dim1=1
  g1[3] = (int)((ne8 & 0xFFFFu) << 16);               // tile_dim0 = ne8 (fits 16b)
  g1[4] = 0;                                          // tile_dim1=0 (1-D), tile_dim2=0
  g1[5] = (int)ne8;                                   // tensor_dim0_stride[31:0]
  g1[6] = 0;                                          // stride hi / dim1_stride lo
  g1[7] = 0;
  i32x4 z4 = {};
  i32x8 z8 = {};
  __builtin_amdgcn_tensor_load_to_lds(g0, g1, z4, z4, z8, 0);
}

// ---------------------------------------------------------------- scatter_mean
__global__ void scatter_kernel(const float* __restrict__ feat,
                               const float* __restrict__ coords,
                               float* __restrict__ gsum,
                               float* __restrict__ cnt) {
  int tid = blockIdx.x * blockDim.x + threadIdx.x;
  if (tid >= BSZ * NPTS) return;
  int b = tid >> 16, n = tid & (NPTS - 1);
  float nc0 = norm_coord(coords[(b * 3 + 0) * NPTS + n]);
  float nc1 = norm_coord(coords[(b * 3 + 1) * NPTS + n]);
  float nc2 = norm_coord(coords[(b * 3 + 2) * NPTS + n]);
  int idx0 = min(max((int)floorf(nc0 + 0.5f), 0), RG - 1);
  int idx1 = min(max((int)floorf(nc1 + 0.5f), 0), RG - 1);
  int idx2 = min(max((int)floorf(nc2 + 0.5f), 0), RG - 1);
  int vox = (idx0 * RG + idx1) * RG + idx2;
  atomicAdd(&cnt[b * R3 + vox], 1.0f);
  float* base = gsum + (size_t)(b * R3 + vox) * CIN_;
#pragma unroll
  for (int c = 0; c < CIN_; ++c)
    atomicAdd(&base[c], feat[(b * CIN_ + c) * NPTS + n]);
}

// sum/count -> mean, written as channel-last bf16 grid (B,R,R,R,CIN)
__global__ void finalize_kernel(const float* __restrict__ gsum,
                                const float* __restrict__ cnt,
                                unsigned short* __restrict__ vin) {
  int tid = blockIdx.x * blockDim.x + threadIdx.x;
  if (tid >= BSZ * R3) return;
  float inv = 1.0f / fmaxf(cnt[tid], 1.0f);
  const float* s = gsum + (size_t)tid * CIN_;
  unsigned short* o = vin + (size_t)tid * CIN_;
#pragma unroll
  for (int c = 0; c < CIN_; ++c) o[c] = f2bf(s[c] * inv);
}

// -------------------------------------------- pack weights into WMMA A layout
// wmp : [mt(4)][lane(32)][e(16)]                    (w_mlp 64x32)
// w1p : [t(27)][mt(4)][lane(32)][e(16)]             (w_conv1 64x32x27)
// w2p : [t(27)][kk(2)][mt(4)][lane(32)][e(16)]      (w_conv2 64x64x27)
__global__ void pack_kernel(const float* __restrict__ wm,
                            const float* __restrict__ w1,
                            const float* __restrict__ w2,
                            unsigned short* __restrict__ wmp,
                            unsigned short* __restrict__ w1p,
                            unsigned short* __restrict__ w2p) {
  int tid = blockIdx.x * blockDim.x + threadIdx.x;
  if (tid < 2048) {
    int i = tid;
    int e = i & 15, L = (i >> 4) & 31, mt = i >> 9;
    int o = mt * 16 + (L & 15), k = kmap(e, L >> 4);
    wmp[i] = f2bf(wm[o * CIN_ + k]);
  } else if (tid < 2048 + 55296) {
    int i = tid - 2048;
    int e = i & 15, L = (i >> 4) & 31, mt = (i >> 9) & 3, t = i >> 11;
    int o = mt * 16 + (L & 15), k = kmap(e, L >> 4);
    w1p[i] = f2bf(w1[(o * CIN_ + k) * 27 + t]);
  } else if (tid < 2048 + 55296 + 110592) {
    int i = tid - (2048 + 55296);
    int e = i & 15, L = (i >> 4) & 31, mt = (i >> 9) & 3, kk = (i >> 11) & 1, t = i >> 12;
    int o = mt * 16 + (L & 15), c = kk * 32 + kmap(e, L >> 4);
    w2p[i] = f2bf(w2[(o * COUT_ + c) * 27 + t]);
  }
}

// --------------------------------------------------- point MLP: relu(W@f + b)
__global__ void mlp_wmma_kernel(const float* __restrict__ feat,
                                const float* __restrict__ bias,
                                const unsigned short* __restrict__ wmp,
                                float* __restrict__ out) {
  int wv = (blockIdx.x * blockDim.x + threadIdx.x) >> 5;
  int lane = threadIdx.x & 31;
  int col = lane & 15, g = lane >> 4;
  int b = wv >> 12;                       // 4096 point-tiles per batch
  int n = ((wv & 4095) << 4) + col;

  v16bf bm;                               // B operand: 32(K) x 16(points)
#pragma unroll
  for (int e = 0; e < 16; ++e) {
    int c = g * 16 + e;                   // B layout: lanes 0-15 K=0..15, 16-31 K=16..31
    bm[e] = (__bf16)feat[(b * CIN_ + c) * NPTS + n];
  }
  v8f acc[4];
#pragma unroll
  for (int mt = 0; mt < 4; ++mt) {
#pragma unroll
    for (int r = 0; r < 8; ++r) acc[mt][r] = bias[mt * 16 + r + 8 * g];
    v16bf am = __builtin_bit_cast(v16bf, *(const u32x8*)(wmp + ((mt * 32 + lane) << 4)));
    acc[mt] = __builtin_amdgcn_wmma_f32_16x16x32_bf16(false, am, false, bm,
                                                      (short)0, acc[mt], false, false);
  }
#pragma unroll
  for (int mt = 0; mt < 4; ++mt)
#pragma unroll
    for (int r = 0; r < 8; ++r) {
      int o = mt * 16 + r + 8 * g;
      out[(b * COUT_ + o) * NPTS + n] = fmaxf(acc[mt][r], 0.0f);
    }
}

// ------------------------------------------------ 3x3x3 conv as implicit GEMM
// Weights staged into LDS once per workgroup via the Tensor Data Mover; all
// A-operand reads served from LDS. 8 waves/block; each wave owns one 16-point
// spatial tile and all 64 output channels.
template <int KCH>
__global__ void conv_wmma_kernel(const unsigned short* __restrict__ in_cl,
                                 const unsigned short* __restrict__ wp,
                                 const float* __restrict__ bias,
                                 unsigned short* __restrict__ out_bf_cl,
                                 float* __restrict__ out_cf,
                                 unsigned short* __restrict__ out_bf_cl2) {
  const int Cin = KCH * 32;
  const unsigned WBYTES = (KCH == 1) ? 110592u : 221184u;
  extern __shared__ unsigned short lds_wp[];

  if ((threadIdx.x >> 5) == 0) {          // wave 0 issues the DMA
    tdm_load_to_lds(wp, (unsigned)(size_t)(void*)lds_wp, WBYTES);
    __builtin_amdgcn_s_wait_tensorcnt(0);
  }
  __syncthreads();

  int wv = (blockIdx.x * blockDim.x + threadIdx.x) >> 5;
  int lane = threadIdx.x & 31;
  int col = lane & 15, g = lane >> 4;
  int xb = wv & 1, y = (wv >> 1) & 31, z = (wv >> 6) & 31, b = wv >> 11;
  int x = xb * 16 + col;

  v8f acc[4];
#pragma unroll
  for (int mt = 0; mt < 4; ++mt)
#pragma unroll
    for (int r = 0; r < 8; ++r) acc[mt][r] = bias[mt * 16 + r + 8 * g];

  const u32x8 zero = {};
#pragma unroll
  for (int t = 0; t < 27; ++t) {
    int dz = t / 9, dy = (t / 3) % 3, dx = t % 3;
    int zz = z + dz - 1, yy = y + dy - 1;
    if (zz < 0 || zz > RG - 1 || yy < 0 || yy > RG - 1) continue;  // wave-uniform
    int xx = x + dx - 1;
    bool ok = (xx >= 0) && (xx < RG);                              // per-lane pad
    int vox = ((b * RG + zz) * RG + yy) * RG + (ok ? xx : 0);
#pragma unroll
    for (int kk = 0; kk < KCH; ++kk) {
      u32x8 braw = *(const u32x8*)(in_cl + (size_t)vox * Cin + kk * 32 + g * 16);
      if (!ok) braw = zero;  // zero padded columns without touching EXEC
      v16bf bm = __builtin_bit_cast(v16bf, braw);
#pragma unroll
      for (int mt = 0; mt < 4; ++mt) {
        v16bf am = __builtin_bit_cast(
            v16bf, *(const u32x8*)(lds_wp + ((((t * KCH + kk) * 4 + mt) * 32 + lane) << 4)));
        acc[mt] = __builtin_amdgcn_wmma_f32_16x16x32_bf16(false, am, false, bm,
                                                          (short)0, acc[mt], false, false);
      }
    }
  }

  int ovox = ((b * RG + z) * RG + y) * RG + x;
  if (KCH == 1) {
#pragma unroll
    for (int mt = 0; mt < 4; ++mt) {
      unsigned int w[4];
#pragma unroll
      for (int p = 0; p < 4; ++p) {
        float f0 = acc[mt][2 * p];     f0 = f0 >= 0.f ? f0 : NEGS * f0;
        float f1 = acc[mt][2 * p + 1]; f1 = f1 >= 0.f ? f1 : NEGS * f1;
        w[p] = (unsigned)f2bf(f0) | ((unsigned)f2bf(f1) << 16);
      }
      uint4 v; v.x = w[0]; v.y = w[1]; v.z = w[2]; v.w = w[3];
      *(uint4*)(out_bf_cl + (size_t)ovox * COUT_ + mt * 16 + g * 8) = v;
    }
  } else {
    int sp = (z * RG + y) * RG + x;
#pragma unroll
    for (int mt = 0; mt < 4; ++mt)
#pragma unroll
      for (int r = 0; r < 8; ++r) {
        float f = acc[mt][r]; f = f >= 0.f ? f : NEGS * f;
        int o = mt * 16 + r + 8 * g;
        out_cf[(size_t)(b * COUT_ + o) * R3 + sp] = f;
        out_bf_cl2[(size_t)ovox * COUT_ + o] = f2bf(f);
      }
  }
}

// ------------------------------------- trilinear sample (bf16 CL grid) + add
__global__ void sample_add_kernel(const float* __restrict__ coords,
                                  const unsigned short* __restrict__ vf_cl,
                                  float* __restrict__ out_pts) {
  int tid = blockIdx.x * blockDim.x + threadIdx.x;
  if (tid >= BSZ * NPTS) return;
  int b = tid >> 16, n = tid & (NPTS - 1);

  float fr[3]; int lo[3], hi[3];
#pragma unroll
  for (int d = 0; d < 3; ++d) {
    float nc = norm_coord(coords[(b * 3 + d) * NPTS + n]);
    int l = min(max((int)floorf(nc), 0), RG - 1);
    lo[d] = l; hi[d] = min(l + 1, RG - 1); fr[d] = nc - (float)l;
  }
  float w[8]; const unsigned short* p[8];
#pragma unroll
  for (int cc = 0; cc < 8; ++cc) {
    int i0 = (cc & 4) ? hi[0] : lo[0];
    int i1 = (cc & 2) ? hi[1] : lo[1];
    int i2 = (cc & 1) ? hi[2] : lo[2];
    float w0 = (cc & 4) ? fr[0] : 1.f - fr[0];
    float w1 = (cc & 2) ? fr[1] : 1.f - fr[1];
    float w2 = (cc & 1) ? fr[2] : 1.f - fr[2];
    w[cc] = w0 * w1 * w2;
    p[cc] = vf_cl + ((size_t)b * R3 + (i0 * RG + i1) * RG + i2) * COUT_;
  }
#pragma unroll
  for (int cb = 0; cb < 4; ++cb) {           // 16 channels per chunk
    float s[16];
#pragma unroll
    for (int e = 0; e < 16; ++e) s[e] = 0.f;
#pragma unroll
    for (int cc = 0; cc < 8; ++cc) {
      u32x8 raw = *(const u32x8*)(p[cc] + cb * 16);
#pragma unroll
      for (int e = 0; e < 8; ++e) {
        unsigned int u = raw[e];
        s[2 * e]     += w[cc] * bf2f((unsigned short)(u & 0xFFFFu));
        s[2 * e + 1] += w[cc] * bf2f((unsigned short)(u >> 16));
      }
    }
#pragma unroll
    for (int e = 0; e < 16; ++e) {
      int c = cb * 16 + e;
      out_pts[(b * COUT_ + c) * NPTS + n] += s[e];
    }
  }
}

// ---------------------------------------------------------------------- launch
extern "C" void kernel_launch(void* const* d_in, const int* in_sizes, int n_in,
                              void* d_out, int out_size, void* d_ws, size_t ws_size,
                              hipStream_t stream) {
  (void)in_sizes; (void)n_in; (void)out_size; (void)ws_size;
  const float* feat   = (const float*)d_in[0];
  const float* coords = (const float*)d_in[1];
  const float* w_mlp  = (const float*)d_in[2];
  const float* b_mlp  = (const float*)d_in[3];
  const float* w1     = (const float*)d_in[4];
  const float* b1     = (const float*)d_in[5];
  const float* w2     = (const float*)d_in[6];
  const float* b2     = (const float*)d_in[7];

  float* out_pts = (float*)d_out;                          // (B,64,N)
  float* out_vox = out_pts + (size_t)BSZ * COUT_ * NPTS;   // (B,64,R^3)

  // workspace layout (peak ~42.8 MB; vf_cl reuses the dead gsum region)
  char* ws = (char*)d_ws;
  float*          gsum = (float*)(ws + 0);                   // 16,777,216 B
  float*          cnt  = (float*)(ws + 16777216);            //    524,288 B
  unsigned short* vin  = (unsigned short*)(ws + 17301504);   //  8,388,608 B
  unsigned short* v1   = (unsigned short*)(ws + 25690112);   // 16,777,216 B
  unsigned short* wmp  = (unsigned short*)(ws + 42467328);   //      4,096 B
  unsigned short* w1p  = (unsigned short*)(ws + 42471424);   //    110,592 B
  unsigned short* w2p  = (unsigned short*)(ws + 42582016);   //    221,184 B
  unsigned short* vfcl = (unsigned short*)(ws + 0);          // 16,777,216 B (reuse)

  // allow >64KB dynamic LDS for the conv kernels (host-side, capture-safe)
  (void)hipFuncSetAttribute((const void*)conv_wmma_kernel<1>,
                            hipFuncAttributeMaxDynamicSharedMemorySize, 110592);
  (void)hipFuncSetAttribute((const void*)conv_wmma_kernel<2>,
                            hipFuncAttributeMaxDynamicSharedMemorySize, 221184);

  (void)hipMemsetAsync(ws, 0, 16777216 + 524288, stream);
  scatter_kernel <<<1024, 256, 0, stream>>>(feat, coords, gsum, cnt);
  finalize_kernel<<< 512, 256, 0, stream>>>(gsum, cnt, vin);
  pack_kernel    <<< 656, 256, 0, stream>>>(w_mlp, w1, w2, wmp, w1p, w2p);
  mlp_wmma_kernel<<<4096, 128, 0, stream>>>(feat, b_mlp, wmp, out_pts);
  conv_wmma_kernel<1><<<1024, 256, 110592, stream>>>(vin, w1p, b1, v1, nullptr, nullptr);
  conv_wmma_kernel<2><<<1024, 256, 221184, stream>>>(v1, w2p, b2, nullptr, out_vox, vfcl);
  sample_add_kernel<<<1024, 256, 0, stream>>>(coords, vfcl, out_pts);
}
